// EncoderBlock_55353538511005
// MI455X (gfx1250) — matrix operand order repaired and checked by reading, not compiled, not run on hardware
//
#include <hip/hip_runtime.h>
#include <hip/hip_bf16.h>
#include <math.h>

// ---------------------------------------------------------------------------
// Types for gfx1250 WMMA (wave32): bf16 16x16x32, fp32 accumulate
// ---------------------------------------------------------------------------
typedef __bf16 bf16_t;
typedef __attribute__((ext_vector_type(16))) __bf16 v16bf;
typedef __attribute__((ext_vector_type(8)))  __bf16 v8bf;
typedef __attribute__((ext_vector_type(8)))  float  v8f;

#define WAVE 32

// Problem constants
#define BB 4
#define SS 2048
#define DD 1024
#define HH 16
#define DH 64
#define MLP 4096
#define BS (BB*SS)   // 8192 rows

// ---------------------------------------------------------------------------
// CDNA5 async global->LDS copy (ASYNCcnt-tracked DMA, no VGPR round trip)
// ---------------------------------------------------------------------------
__device__ __forceinline__ unsigned lds_offset(void* p) {
  // __shared__ pointers are generic; addrspace(3) cast yields the 0-based LDS
  // offset that GLOBAL_LOAD_ASYNC_TO_LDS's VDST operand expects.
  return (unsigned)(size_t)(__attribute__((address_space(3))) char*)(char*)p;
}

__device__ __forceinline__ void async_load_b128(unsigned lds_off,
                                                const void* gaddr) {
  asm volatile("global_load_async_to_lds_b128 %0, %1, off"
               :: "v"(lds_off), "v"((unsigned long long)(size_t)gaddr)
               : "memory");
}

__device__ __forceinline__ void wait_async0() {
  asm volatile("s_wait_asynccnt 0" ::: "memory");
}

// ---------------------------------------------------------------------------
// Fragment loaders (layouts per CDNA5 ISA 7.12.2, wave32)
// A 16x32 bf16: lane L -> row M=L%16; half=L/16; v0..3 hold K=half*8+0..7,
//               v4..7 hold K=16+half*8+0..7 (2 bf16 per VGPR, LE packed)
// B 32x16 bf16 (read from an N-major "transposed" tile bt[n][k], ld elems):
//               lane L -> col N=L%16; same K order as A (half*8+0..7, 16+half*8+0..7)
// C 16x16 f32:  acc[r] at row M=r+8*(L/16), col N=L%16
// ---------------------------------------------------------------------------
__device__ __forceinline__ v16bf load_a_frag(const bf16_t* base, int ld) {
  int lane = threadIdx.x & (WAVE - 1);
  int m    = lane & 15;
  int half = lane >> 4;
  const bf16_t* row = base + m * ld;
  union { v16bf v; v8bf h[2]; } u;
  u.h[0] = *(const v8bf*)(row + half * 8);
  u.h[1] = *(const v8bf*)(row + 16 + half * 8);
  return u.v;
}

__device__ __forceinline__ v16bf load_b_frag_t(const bf16_t* bt_base, int ld) {
  int lane = threadIdx.x & (WAVE - 1);
  int n    = lane & 15;
  int half = lane >> 4;
  const bf16_t* row = bt_base + n * ld + half * 8;
  union { v16bf v; v8bf h[2]; } u;
  u.h[0] = *(const v8bf*)(row);
  u.h[1] = *(const v8bf*)(row + 16);
  return u.v;
}

// ---------------------------------------------------------------------------
// fp32 -> bf16 weight conversion
// ---------------------------------------------------------------------------
__global__ void cvt_bf16_kernel(const float* __restrict__ in,
                                bf16_t* __restrict__ out, int n) {
  int i = blockIdx.x * 256 + threadIdx.x;
  if (i < n) out[i] = (bf16_t)in[i];
}

// ---------------------------------------------------------------------------
// LayerNorm: one workgroup (256 thr) per row of D=1024, bf16 output
// ---------------------------------------------------------------------------
__global__ __launch_bounds__(256) void ln_kernel(const float* __restrict__ x,
                                                 const float* __restrict__ g,
                                                 const float* __restrict__ b,
                                                 bf16_t* __restrict__ out) {
  int row = blockIdx.x;
  int tid = threadIdx.x;
  const float* xr = x + (size_t)row * DD;
  float v[4], s = 0.f, ss = 0.f;
#pragma unroll
  for (int i = 0; i < 4; ++i) {
    float t = xr[tid + 256 * i];
    v[i] = t; s += t; ss += t * t;
  }
#pragma unroll
  for (int off = 1; off < WAVE; off <<= 1) {
    s  += __shfl_xor(s,  off, WAVE);
    ss += __shfl_xor(ss, off, WAVE);
  }
  __shared__ float rs[8], rss[8];
  int wave = tid >> 5, lane = tid & 31;
  if (lane == 0) { rs[wave] = s; rss[wave] = ss; }
  __syncthreads();
  s = 0.f; ss = 0.f;
#pragma unroll
  for (int w = 0; w < 8; ++w) { s += rs[w]; ss += rss[w]; }
  float mu   = s * (1.0f / DD);
  float var  = ss * (1.0f / DD) - mu * mu;
  float rstd = rsqrtf(var + 1e-6f);
#pragma unroll
  for (int i = 0; i < 4; ++i) {
    int c = tid + 256 * i;
    out[(size_t)row * DD + c] = (bf16_t)((v[i] - mu) * rstd * g[c] + b[c]);
  }
}

// ---------------------------------------------------------------------------
// Generic WMMA GEMM: C[M,N] = act(A[M,K]@B[K,N] + bias) (+ resid)
// 128x128 tile per WG (8 waves as 4Mx2N grid, each wave 32Mx64N -> 8 WMMA per
// 32-wide K step). Double-buffered LDS, one barrier per K step, A tile via
// async global->LDS DMA, B tile transposed through VGPRs.
// out_mode: 0 = bf16 store; 1 = bf16 store with exact GELU; 2 = f32 + resid
// ---------------------------------------------------------------------------
__global__ __launch_bounds__(256) void gemm_wmma_kernel(
    const bf16_t* __restrict__ A, const bf16_t* __restrict__ Bm,
    const float* __restrict__ bias, const float* __restrict__ resid,
    void* __restrict__ outp, int M, int N, int K, int out_mode) {
  __shared__ __align__(16) bf16_t sA[2][128 * 32];    // 2 x 8 KB
  __shared__ __align__(16) bf16_t sBt[2][128 * 32];   // 2 x 8 KB, N-major
  int tid  = threadIdx.x;
  int wave = tid >> 5;
  int lane = tid & 31;
  int half = lane >> 4;
  int ln   = lane & 15;
  int m0 = blockIdx.y * 128;
  int n0 = blockIdx.x * 128;
  int wm = (wave & 3) * 32;     // wave's M offset inside tile
  int wn = (wave >> 2) * 64;    // wave's N offset inside tile

  // per-thread tile-load coordinates
  int ar = tid >> 1;            // A: 0..127 row
  int ac = (tid & 1) * 16;      // A: 0 or 16 col
  int bk = tid >> 3;            // B: 0..31 k
  int bc = (tid & 7) * 16;      // B: 0..112 n chunk

  unsigned sA0 = lds_offset(&sA[0][0]);
  unsigned sA_off[2];
  sA_off[0] = sA0 + (unsigned)(ar * 32 + ac) * 2u;
  sA_off[1] = sA_off[0] + 128u * 32u * 2u;

  v8f acc[2][4] = {};

  // prologue: issue tile 0 into buffer 0
  {
    const bf16_t* srcA = A + (size_t)(m0 + ar) * K + ac;
    async_load_b128(sA_off[0],      srcA);
    async_load_b128(sA_off[0] + 16, srcA + 8);
    union { v8bf v; bf16_t e[8]; } u0, u1;
    u0.v = *(const v8bf*)(Bm + (size_t)bk * N + n0 + bc);
    u1.v = *(const v8bf*)(Bm + (size_t)bk * N + n0 + bc + 8);
#pragma unroll
    for (int j = 0; j < 8; ++j) {
      sBt[0][(bc + j) * 32 + bk]     = u0.e[j];
      sBt[0][(bc + 8 + j) * 32 + bk] = u1.e[j];
    }
  }

  for (int kk = 0; kk < K; kk += 32) {
    int buf = (kk >> 5) & 1;
    wait_async0();        // our async DMA into `buf` is complete
    __syncthreads();      // everyone's DMA + transpose stores into `buf` done

    if (kk + 32 < K) {    // issue next tile into the other buffer
      int nb = buf ^ 1;
      const bf16_t* srcA = A + (size_t)(m0 + ar) * K + (kk + 32) + ac;
      async_load_b128(sA_off[nb],      srcA);
      async_load_b128(sA_off[nb] + 16, srcA + 8);
      union { v8bf v; bf16_t e[8]; } u0, u1;
      u0.v = *(const v8bf*)(Bm + (size_t)(kk + 32 + bk) * N + n0 + bc);
      u1.v = *(const v8bf*)(Bm + (size_t)(kk + 32 + bk) * N + n0 + bc + 8);
#pragma unroll
      for (int j = 0; j < 8; ++j) {
        sBt[nb][(bc + j) * 32 + bk]     = u0.e[j];
        sBt[nb][(bc + 8 + j) * 32 + bk] = u1.e[j];
      }
    }

    v16bf af[2];
#pragma unroll
    for (int a = 0; a < 2; ++a)
      af[a] = load_a_frag(&sA[buf][(wm + a * 16) * 32], 32);
#pragma unroll
    for (int t = 0; t < 4; ++t) {
      v16bf b = load_b_frag_t(&sBt[buf][(wn + t * 16) * 32], 32);
#pragma unroll
      for (int a = 0; a < 2; ++a)
        acc[a][t] = __builtin_amdgcn_wmma_f32_16x16x32_bf16(
            false, af[a], false, b, (short)0, acc[a][t], false, false);
    }
  }

  // epilogue
#pragma unroll
  for (int a = 0; a < 2; ++a) {
#pragma unroll
    for (int t = 0; t < 4; ++t) {
      union { v8f v; float f[8]; } u; u.v = acc[a][t];
#pragma unroll
      for (int r = 0; r < 8; ++r) {
        int row = m0 + wm + a * 16 + r + 8 * half;
        int col = n0 + wn + t * 16 + ln;
        float val = u.f[r] + bias[col];
        size_t idx = (size_t)row * N + col;
        if (out_mode == 2) {
          ((float*)outp)[idx] = val + resid[idx];
        } else {
          if (out_mode == 1)
            val = 0.5f * val * (1.0f + erff(val * 0.70710678118654752f));
          ((bf16_t*)outp)[idx] = (bf16_t)val;
        }
      }
    }
  }
}

// ---------------------------------------------------------------------------
// Flash attention: grid (S/128, B*H); 256 thr = 8 waves, 16 q-rows per wave.
// Q/K/V bf16 in [BS, D] layout, head h at column h*64. Online softmax.
// ---------------------------------------------------------------------------
__global__ __launch_bounds__(256) void attn_kernel(
    const bf16_t* __restrict__ Q, const bf16_t* __restrict__ Km,
    const bf16_t* __restrict__ Vm, bf16_t* __restrict__ O) {
  int bh = blockIdx.y;
  int b  = bh >> 4;
  int h  = bh & 15;
  int q0 = blockIdx.x * 128;
  int tid = threadIdx.x, wave = tid >> 5, lane = tid & 31;
  int half = lane >> 4, ln = lane & 15;
  const size_t rowbase = (size_t)b * SS;
  const int colh = h * DH;

  __shared__ __align__(16) bf16_t sK[32 * 64];      // 4 KB, key-major
  __shared__ __align__(16) bf16_t sVt[64 * 32];     // 4 KB, dh-major
  __shared__ __align__(16) bf16_t sP[8][16 * 32];   // 8 KB, per-wave P tile

  // Q fragments for this wave's 16 rows (loaded once): dh chunks 0..31, 32..63
  int qrow = q0 + wave * 16 + ln;
  const bf16_t* qp = Q + (rowbase + qrow) * DD + colh;
  v16bf qa[2];
#pragma unroll
  for (int d = 0; d < 2; ++d) {
    union { v16bf v; v8bf h2[2]; } u;
    u.h2[0] = *(const v8bf*)(qp + d * 32 + half * 8);
    u.h2[1] = *(const v8bf*)(qp + d * 32 + 16 + half * 8);
    qa[d] = u.v;
  }

  v8f o[4] = {};
  float mrow[8], lrow[8];
#pragma unroll
  for (int r = 0; r < 8; ++r) { mrow[r] = -1e30f; lrow[r] = 0.f; }

  for (int kt = 0; kt < SS; kt += 32) {
    __syncthreads();
    { // cooperative load of 32x64 K and V tiles; V stored transposed
      int r = tid >> 3;          // 0..31 key within tile
      int c = (tid & 7) * 8;     // 0..56 dh
      *(v8bf*)(sK + r * 64 + c) =
          *(const v8bf*)(Km + (rowbase + kt + r) * DD + colh + c);
      union { v8bf v; bf16_t e[8]; } uv;
      uv.v = *(const v8bf*)(Vm + (rowbase + kt + r) * DD + colh + c);
#pragma unroll
      for (int j = 0; j < 8; ++j) sVt[(c + j) * 32 + r] = uv.e[j];
    }
    __syncthreads();

    // scores S = Q K^T : two 16-key subtiles, each accumulated over dh (2x32)
    v8f sc[2] = {};
#pragma unroll
    for (int nt = 0; nt < 2; ++nt) {
#pragma unroll
      for (int d = 0; d < 2; ++d) {
        // B frag of K^T: lane col N = key (nt*16+ln), K dim = dh; A-matching order
        union { v16bf v; v8bf h2[2]; } u;
        const bf16_t* kp = sK + (nt * 16 + ln) * 64 + d * 32 + half * 8;
        u.h2[0] = *(const v8bf*)(kp);
        u.h2[1] = *(const v8bf*)(kp + 16);
        sc[nt] = __builtin_amdgcn_wmma_f32_16x16x32_bf16(
            false, qa[d], false, u.v, (short)0, sc[nt], false, false);
      }
    }

    // online softmax over this 32-key tile
    union { v8f v; float f[8]; } u0, u1; u0.v = sc[0]; u1.v = sc[1];
    float p0a[8], p1a[8];
#pragma unroll
    for (int r = 0; r < 8; ++r) {
      float v0 = u0.f[r] * 0.125f;     // 1/sqrt(64)
      float v1 = u1.f[r] * 0.125f;
      float mx = fmaxf(v0, v1);
#pragma unroll
      for (int off = 1; off < 16; off <<= 1)
        mx = fmaxf(mx, __shfl_xor(mx, off, WAVE));
      float mnew = fmaxf(mrow[r], mx);
      float corr = expf(mrow[r] - mnew);
      float p0 = expf(v0 - mnew);
      float p1 = expf(v1 - mnew);
      float ps = p0 + p1;
#pragma unroll
      for (int off = 1; off < 16; off <<= 1)
        ps += __shfl_xor(ps, off, WAVE);
      lrow[r] = lrow[r] * corr + ps;
      mrow[r] = mnew;
#pragma unroll
      for (int t = 0; t < 4; ++t) o[t][r] *= corr;
      p0a[r] = p0; p1a[r] = p1;
    }

    // re-swizzle P (C layout) -> A layout via per-wave LDS tile
    bf16_t* pw = sP[wave];
#pragma unroll
    for (int r = 0; r < 8; ++r) {
      int row = r + 8 * half;
      pw[row * 32 + ln]      = (bf16_t)p0a[r];
      pw[row * 32 + 16 + ln] = (bf16_t)p1a[r];
    }
    union { v16bf v; v8bf h2[2]; } ua;
    const bf16_t* pr = pw + ln * 32;
    ua.h2[0] = *(const v8bf*)(pr + half * 8);
    ua.h2[1] = *(const v8bf*)(pr + 16 + half * 8);

    // O += P V : K dim = 32 keys, 4 dh subtiles of 16
#pragma unroll
    for (int t = 0; t < 4; ++t) {
      union { v16bf v; v8bf h2[2]; } ub;
      const bf16_t* vp = sVt + (t * 16 + ln) * 32 + half * 8;
      ub.h2[0] = *(const v8bf*)(vp);
      ub.h2[1] = *(const v8bf*)(vp + 16);
      o[t] = __builtin_amdgcn_wmma_f32_16x16x32_bf16(
          false, ua.v, false, ub.v, (short)0, o[t], false, false);
    }
  }

  // finalize: divide by softmax denominator, write bf16
#pragma unroll
  for (int t = 0; t < 4; ++t) {
    union { v8f v; float f[8]; } u; u.v = o[t];
#pragma unroll
    for (int r = 0; r < 8; ++r) {
      int row = q0 + wave * 16 + r + 8 * half;
      float val = u.f[r] / lrow[r];
      O[(rowbase + row) * DD + colh + t * 16 + ln] = (bf16_t)val;
    }
  }
}

// ---------------------------------------------------------------------------
// Host-side launch
// ---------------------------------------------------------------------------
extern "C" void kernel_launch(void* const* d_in, const int* in_sizes, int n_in,
                              void* d_out, int out_size, void* d_ws, size_t ws_size,
                              hipStream_t stream) {
  (void)in_sizes; (void)n_in; (void)out_size; (void)ws_size;
  const float* x     = (const float*)d_in[0];
  const float* Wq    = (const float*)d_in[1];
  const float* bq    = (const float*)d_in[2];
  const float* Wk    = (const float*)d_in[3];
  const float* bk    = (const float*)d_in[4];
  const float* Wv    = (const float*)d_in[5];
  const float* bv    = (const float*)d_in[6];
  const float* Wo    = (const float*)d_in[7];
  const float* bo    = (const float*)d_in[8];
  const float* ln1_g = (const float*)d_in[9];
  const float* ln1_b = (const float*)d_in[10];
  const float* ln2_g = (const float*)d_in[11];
  const float* ln2_b = (const float*)d_in[12];
  const float* W1    = (const float*)d_in[13];
  const float* b1    = (const float*)d_in[14];
  const float* W2    = (const float*)d_in[15];
  const float* b2    = (const float*)d_in[16];

  const size_t MB = 1024ull * 1024ull;
  char* ws = (char*)d_ws;
  bf16_t* xn    = (bf16_t*)(ws + 0 * MB);     // 16 MB  (reused for xn2)
  bf16_t* Qb    = (bf16_t*)(ws + 16 * MB);    // 16 MB
  bf16_t* Kb    = (bf16_t*)(ws + 32 * MB);    // 16 MB
  bf16_t* Vb    = (bf16_t*)(ws + 48 * MB);    // 16 MB
  bf16_t* Rb    = (bf16_t*)(ws + 64 * MB);    // 16 MB  attn output pre-Wo
  float*  x1    = (float*) (ws + 80 * MB);    // 32 MB  residual after attn
  bf16_t* Mb    = (bf16_t*)(ws + 112 * MB);   // 64 MB  MLP hidden
  bf16_t* Wq_h  = (bf16_t*)(ws + 176 * MB);   // 2 MB each
  bf16_t* Wk_h  = (bf16_t*)(ws + 178 * MB);
  bf16_t* Wv_h  = (bf16_t*)(ws + 180 * MB);
  bf16_t* Wo_h  = (bf16_t*)(ws + 182 * MB);
  bf16_t* W1_h  = (bf16_t*)(ws + 184 * MB);   // 8 MB
  bf16_t* W2_h  = (bf16_t*)(ws + 192 * MB);   // 8 MB

  // 1) weights -> bf16
  {
    int nDD = DD * DD, nDM = DD * MLP;
    int gDD = (nDD + 255) / 256, gDM = (nDM + 255) / 256;
    cvt_bf16_kernel<<<gDD, 256, 0, stream>>>(Wq, Wq_h, nDD);
    cvt_bf16_kernel<<<gDD, 256, 0, stream>>>(Wk, Wk_h, nDD);
    cvt_bf16_kernel<<<gDD, 256, 0, stream>>>(Wv, Wv_h, nDD);
    cvt_bf16_kernel<<<gDD, 256, 0, stream>>>(Wo, Wo_h, nDD);
    cvt_bf16_kernel<<<gDM, 256, 0, stream>>>(W1, W1_h, nDM);
    cvt_bf16_kernel<<<gDM, 256, 0, stream>>>(W2, W2_h, nDM);
  }

  // 2) LN1
  ln_kernel<<<BS, 256, 0, stream>>>(x, ln1_g, ln1_b, xn);

  // 3) Q/K/V projections (bf16 out)
  dim3 gP(DD / 128, BS / 128);
  gemm_wmma_kernel<<<gP, 256, 0, stream>>>(xn, Wq_h, bq, nullptr, Qb, BS, DD, DD, 0);
  gemm_wmma_kernel<<<gP, 256, 0, stream>>>(xn, Wk_h, bk, nullptr, Kb, BS, DD, DD, 0);
  gemm_wmma_kernel<<<gP, 256, 0, stream>>>(xn, Wv_h, bv, nullptr, Vb, BS, DD, DD, 0);

  // 4) attention
  dim3 gA(SS / 128, BB * HH);
  attn_kernel<<<gA, 256, 0, stream>>>(Qb, Kb, Vb, Rb);

  // 5) output projection + residual (fp32 out)
  gemm_wmma_kernel<<<gP, 256, 0, stream>>>(Rb, Wo_h, bo, x, x1, BS, DD, DD, 2);

  // 6) LN2
  ln_kernel<<<BS, 256, 0, stream>>>(x1, ln2_g, ln2_b, xn);

  // 7) MLP up + exact GELU (bf16 out)
  dim3 gU(MLP / 128, BS / 128);
  gemm_wmma_kernel<<<gU, 256, 0, stream>>>(xn, W1_h, b1, nullptr, Mb, BS, MLP, DD, 1);

  // 8) MLP down + residual (fp32 out to d_out)
  gemm_wmma_kernel<<<gP, 256, 0, stream>>>(Mb, W2_h, b2, x1, d_out, BS, DD, MLP, 2);
}
